// Model_34368328302603
// MI455X (gfx1250) — matrix operand order repaired and checked by reading, not compiled, not run on hardware
//
#include <hip/hip_runtime.h>
#include <math.h>

#define BB   16
#define SS   512
#define CC   64
#define DD   128
#define HH   8
#define PP   96
#define DHH  16
#define DFFD 256
#define BC   (BB*CC)   // 1024 rows for all projections

typedef __attribute__((ext_vector_type(16))) _Float16 v16h;
typedef __attribute__((ext_vector_type(8)))  _Float16 v8h;
typedef __attribute__((ext_vector_type(8)))  float    v8f;

// ---------------------------------------------------------------------------
// h_last[b,c,d] = Wconv[c,d,0]*x[b,S-2,c] + Wconv[c,d,1]*x[b,S-1,c] + bconv[c,d]
// (third conv tap multiplies the zero pad; only the last output position is used)
// ---------------------------------------------------------------------------
__global__ void conv_hlast_kernel(const float* __restrict__ x_enc,
                                  const float* __restrict__ Wconv,
                                  const float* __restrict__ bconv,
                                  float* __restrict__ hl) {
  int i = blockIdx.x * blockDim.x + threadIdx.x;       // B*C*D
  if (i >= BB*CC*DD) return;
  int d = i % DD;
  int c = (i / DD) % CC;
  int b = i / (CC*DD);
  float x0 = x_enc[((size_t)b*SS + (SS-2))*CC + c];
  float x1 = x_enc[((size_t)b*SS + (SS-1))*CC + c];
  const float* w = Wconv + ((size_t)c*DD + d)*3;
  hl[i] = w[0]*x0 + w[1]*x1 + bconv[(size_t)c*DD + d];
}

// ---------------------------------------------------------------------------
// WMMA GEMM: C[1024,N] = act(A[1024,K] @ W[K,N] + bias[N])
// Workgroup = 256 threads = 8 waves. Each workgroup owns one 16-wide column
// panel of W, staged once into LDS as f16 (col-major so each lane's 16
// contiguous K-values are two b128 LDS loads). Each wave computes a 16x16
// tile; 8 waves cover 128 rows -> grid (N/16, 1024/128).
// K templated (128/256) so the loop fully unrolls into back-to-back v_wmma.
// Fragment packing used here (wave32; contiguous k order, same for A and B):
//   A (16x32 f16): g=lane>>4, row=lane&15; element e -> K = 16g + e
//     (pairs e,e+1 contiguous in memory -> float2 loads)
//   B (32x16 f16): col=lane&15; element e -> K = 16g + e
//   D (16x16 f32): vgpr r -> row = r + 8g, col = lane&15
// ACT: 0=none, 1=relu, 2=sigmoid
// ---------------------------------------------------------------------------
template<int K, int ACT>
__global__ void __launch_bounds__(256)
wmma_gemm_kernel(const float* __restrict__ A, const float* __restrict__ W,
                 const float* __restrict__ bias, float* __restrict__ C, int N) {
  __shared__ _Float16 Wl[16][K];          // [col][k] panel, f16
  const int tid   = threadIdx.x;
  const int lane  = tid & 31;
  const int wave  = tid >> 5;             // 0..7
  const int g     = lane >> 4;
  const int mr    = lane & 15;
  const int tileN = blockIdx.x * 16;
  const int tileM = blockIdx.y * 128 + wave * 16;

  const float* Arow = A + (size_t)(tileM + mr) * K;
  __builtin_prefetch(Arow, 0, 1);         // global_prefetch_b8: warm A while staging W

  // stage W panel (K x 16 floats) -> Wl[col][k] as f16; 16 consecutive tids
  // read 16 consecutive floats of one W row (coalesced).
#pragma unroll
  for (int i = tid; i < K * 16; i += 256) {
    int k = i >> 4; int col = i & 15;
    Wl[col][k] = (_Float16)W[(size_t)k * N + tileN + col];
  }
  __syncthreads();

  v8f acc = {};
#pragma unroll
  for (int k0 = 0; k0 < K; k0 += 32) {
    v16h a;
#pragma unroll
    for (int e = 0; e < 16; e += 2) {
      int ka = k0 + 16*g + e;
      float2 f2 = *(const float2*)(Arow + ka);
      a[e]     = (_Float16)f2.x;
      a[e + 1] = (_Float16)f2.y;
    }
    const _Float16* bp = &Wl[mr][k0 + 16*g];   // 16 contiguous halves, 32B aligned
    v8h b0 = *(const v8h*)bp;
    v8h b1 = *(const v8h*)(bp + 8);
    v16h b;
#pragma unroll
    for (int e = 0; e < 8; ++e) { b[e] = b0[e]; b[e + 8] = b1[e]; }
    acc = __builtin_amdgcn_wmma_f32_16x16x32_f16(
        /*neg_a=*/false, a, /*neg_b=*/false, b,
        /*c_mod=*/(short)0, acc, /*reuse_a=*/false, /*reuse_b=*/false);
  }

  float bn = bias[tileN + mr];
#pragma unroll
  for (int r = 0; r < 8; ++r) {
    float v = acc[r] + bn;
    if (ACT == 1)      v = fmaxf(v, 0.0f);
    else if (ACT == 2) v = 1.0f / (1.0f + __expf(-v));
    C[(size_t)(tileM + r + 8*g) * N + tileN + mr] = v;
  }
}

// ---------------------------------------------------------------------------
// Cross-channel attention: one block per (b,h), 64 threads (one per query c).
// q/k/v laid out (B,C,D) with head slice h*16..h*16+15.
// ---------------------------------------------------------------------------
__global__ void xattn_kernel(const float* __restrict__ qb,
                             const float* __restrict__ kb,
                             const float* __restrict__ vb,
                             float* __restrict__ out) {
  int bh = blockIdx.x; int b = bh >> 3; int h = bh & 7;
  int cq = threadIdx.x;                       // query channel
  __shared__ float Ks[CC][DHH+1];
  __shared__ float Vs[CC][DHH+1];
  __shared__ float Sc[CC][CC+1];
  size_t base = ((size_t)b*CC + cq)*DD + h*DHH;
  float q[DHH];
#pragma unroll
  for (int t = 0; t < DHH; ++t) {
    Ks[cq][t] = kb[base + t];
    Vs[cq][t] = vb[base + t];
    q[t]      = qb[base + t];
  }
  __syncthreads();
  float mx = -3.4e38f;
  for (int k = 0; k < CC; ++k) {
    float d = 0.0f;
#pragma unroll
    for (int t = 0; t < DHH; ++t) d += q[t] * Ks[k][t];
    d *= 0.25f;                               // 1/sqrt(16)
    Sc[cq][k] = d;
    mx = fmaxf(mx, d);
  }
  float sum = 0.0f;
  for (int k = 0; k < CC; ++k) { float e = __expf(Sc[cq][k]-mx); Sc[cq][k] = e; sum += e; }
  float inv = 1.0f / sum;
  float o[DHH] = {};
  for (int k = 0; k < CC; ++k) {
    float w = Sc[cq][k] * inv;
#pragma unroll
    for (int t = 0; t < DHH; ++t) o[t] += w * Vs[k][t];
  }
#pragma unroll
  for (int t = 0; t < DHH; ++t) out[base + t] = o[t];
}

// inp = concat(h_last, catt) along feature dim -> (B*C, 256)
__global__ void concat_kernel(const float* __restrict__ a, const float* __restrict__ b,
                              float* __restrict__ inp) {
  int i = blockIdx.x * blockDim.x + threadIdx.x;   // BC*2D
  if (i >= BC*2*DD) return;
  int d = i % (2*DD); int row = i / (2*DD);
  inp[i] = (d < DD) ? a[(size_t)row*DD + d] : b[(size_t)row*DD + d - DD];
}

__global__ void init_kimax_kernel(int* kimax) {
  if (threadIdx.x < CC) kimax[threadIdx.x] = 0;
}

// ki = floor(sigmoid(m1 . Wm2 + bm2) * 63); kimax[c] = max over b
__global__ void ki_kernel(const float* __restrict__ m1, const float* __restrict__ Wm2,
                          const float* __restrict__ bm2, int* __restrict__ kimax) {
  int row = blockIdx.x * blockDim.x + threadIdx.x;  // b*C + c
  if (row >= BC) return;
  float acc = bm2[0];
  for (int t = 0; t < DD; ++t) acc += m1[(size_t)row*DD + t] * Wm2[t];
  float sg = 1.0f / (1.0f + __expf(-acc));
  int ki = (int)floorf(sg * (float)(CC-1));
  atomicMax(&kimax[row % CC], ki);
}

// top-k(63) via exact rank (JAX tie-break: equal values -> lower index first)
__global__ void topk_kernel(const float* __restrict__ si, int* __restrict__ idxb) {
  int row = blockIdx.x;          // b*C + c
  int j = threadIdx.x;           // 0..63
  __shared__ float s[CC];
  s[j] = si[(size_t)row*CC + j];
  __syncthreads();
  float vj = s[j];
  int rank = 0;
  for (int i = 0; i < CC; ++i) {
    float vi = s[i];
    rank += (vi > vj) || (vi == vj && i < j);
  }
  idxb[(size_t)row*CC + rank] = j;   // ranks 0..62 are the top-63 indices
}

// ---------------------------------------------------------------------------
// Gather attention: per (b,c), 64 tokens = [self, top-63 gathered channels];
// keys/values are pre-projected rows of h_last (hk, hv) -> pure gather.
// ---------------------------------------------------------------------------
__global__ void gather_attn_kernel(const float* __restrict__ lq,
                                   const float* __restrict__ hk,
                                   const float* __restrict__ hv,
                                   const int* __restrict__ idxb,
                                   const int* __restrict__ kimax,
                                   float* __restrict__ out) {
  int row = blockIdx.x; int b = row >> 6; int c = row & 63;
  int l = threadIdx.x;                      // token slot 0..63
  __shared__ float sc[HH][CC];
  __shared__ float aw[HH][CC];
  __shared__ int   src[CC];
  int s_ = (l == 0) ? c : idxb[(size_t)row*CC + (l-1)];
  src[l] = s_;
  bool ok = (l == 0) || ((l-1) < kimax[c]);
  const float* krow = hk + ((size_t)b*CC + s_)*DD;
  const float* qrow = lq + (size_t)row*DD;
#pragma unroll
  for (int h = 0; h < HH; ++h) {
    float d = 0.0f;
#pragma unroll
    for (int t = 0; t < DHH; ++t) d += qrow[h*DHH+t] * krow[h*DHH+t];
    sc[h][l] = ok ? d * 0.25f : -1.0e9f;
  }
  __syncthreads();
  if (l < HH) {
    float mx = -3.4e38f;
    for (int i = 0; i < CC; ++i) mx = fmaxf(mx, sc[l][i]);
    float sum = 0.0f;
    for (int i = 0; i < CC; ++i) { float e = __expf(sc[l][i]-mx); aw[l][i] = e; sum += e; }
    float inv = 1.0f / sum;
    for (int i = 0; i < CC; ++i) aw[l][i] *= inv;
  }
  __syncthreads();
#pragma unroll
  for (int rep = 0; rep < 2; ++rep) {
    int d = l + rep*CC; int h = d >> 4;
    float acc = 0.0f;
    for (int i = 0; i < CC; ++i)
      acc += aw[h][i] * hv[((size_t)b*CC + src[i])*DD + d];
    out[(size_t)row*DD + d] = acc;
  }
}

// out = LN(x + r) * g + be, per row of 128
__global__ void ln_res_kernel(const float* __restrict__ x, const float* __restrict__ r,
                              const float* __restrict__ g, const float* __restrict__ be,
                              float* __restrict__ out) {
  int row = blockIdx.x; int t = threadIdx.x;
  __shared__ float red[DD];
  float v = x[(size_t)row*DD + t] + r[(size_t)row*DD + t];
  red[t] = v; __syncthreads();
  for (int s = DD/2; s > 0; s >>= 1) { if (t < s) red[t] += red[t+s]; __syncthreads(); }
  float m = red[0] / (float)DD;
  __syncthreads();
  float dv = v - m;
  red[t] = dv*dv; __syncthreads();
  for (int s = DD/2; s > 0; s >>= 1) { if (t < s) red[t] += red[t+s]; __syncthreads(); }
  float var = red[0] / (float)DD;
  out[(size_t)row*DD + t] = dv * rsqrtf(var + 1e-5f) * g[t] + be[t];
}

// out[b,p] = mean over c of y[b,c,p]
__global__ void mean_kernel(const float* __restrict__ y, float* __restrict__ out) {
  int i = blockIdx.x * blockDim.x + threadIdx.x;   // B*P
  if (i >= BB*PP) return;
  int b = i / PP; int p = i % PP;
  float s = 0.0f;
  for (int c = 0; c < CC; ++c) s += y[((size_t)b*CC + c)*PP + p];
  out[i] = s * (1.0f/(float)CC);
}

// ---------------------------------------------------------------------------
extern "C" void kernel_launch(void* const* d_in, const int* in_sizes, int n_in,
                              void* d_out, int out_size, void* d_ws, size_t ws_size,
                              hipStream_t stream) {
  (void)in_sizes; (void)n_in; (void)out_size; (void)ws_size;
  const float* x_enc = (const float*)d_in[0];
  const float* Wconv = (const float*)d_in[4];  const float* bconv = (const float*)d_in[5];
  const float* Wq  = (const float*)d_in[6];    const float* bq  = (const float*)d_in[7];
  const float* Wk  = (const float*)d_in[8];    const float* bk  = (const float*)d_in[9];
  const float* Wv  = (const float*)d_in[10];   const float* bv  = (const float*)d_in[11];
  const float* Wo  = (const float*)d_in[12];   const float* bo  = (const float*)d_in[13];
  const float* Wm1 = (const float*)d_in[14];   const float* bm1 = (const float*)d_in[15];
  const float* Wm2 = (const float*)d_in[16];   const float* bm2 = (const float*)d_in[17];
  const float* Wc  = (const float*)d_in[18];   const float* bc  = (const float*)d_in[19];
  const float* Wlq = (const float*)d_in[20];   const float* blq = (const float*)d_in[21];
  const float* Wlk = (const float*)d_in[22];   const float* blk_b = (const float*)d_in[23];
  const float* Wlv = (const float*)d_in[24];   const float* blv = (const float*)d_in[25];
  const float* Wlo = (const float*)d_in[26];   const float* blo = (const float*)d_in[27];
  const float* g1  = (const float*)d_in[28];   const float* be1 = (const float*)d_in[29];
  const float* Wf1 = (const float*)d_in[30];   const float* bf1 = (const float*)d_in[31];
  const float* Wf2 = (const float*)d_in[32];   const float* bf2 = (const float*)d_in[33];
  const float* g2  = (const float*)d_in[34];   const float* be2 = (const float*)d_in[35];
  const float* Wp  = (const float*)d_in[36];   const float* bp  = (const float*)d_in[37];
  float* out = (float*)d_out;

  float* ws   = (float*)d_ws;
  float* hl   = ws;              // (B*C,128)
  float* qb   = hl   + 131072;
  float* kb   = qb   + 131072;
  float* vb   = kb   + 131072;
  float* xo   = vb   + 131072;   // cross-attn output
  float* catt = xo   + 131072;
  float* inp  = catt + 131072;   // (B*C,256)
  float* m1   = inp  + 262144;
  float* si   = m1   + 131072;   // (B*C,64)
  float* lqb  = si   + 65536;
  float* hkb  = lqb  + 131072;
  float* hvb  = hkb  + 131072;
  float* a2   = hvb  + 131072;   // gather-attn pre-Wlo
  float* at0  = a2   + 131072;
  float* t0   = at0  + 131072;
  float* f1   = t0   + 131072;   // (B*C,256)
  float* f2   = f1   + 262144;
  float* zb   = f2   + 131072;
  float* yb   = zb   + 131072;   // (B*C,96)
  int*   idxb = (int*)(yb + 98304);   // (B*C,64) ranks
  int*   kimax= idxb + 65536;         // (64)

  dim3 tpb(256);
  dim3 gDD(DD/16,   BC/128);   // N=128
  dim3 gFF(DFFD/16, BC/128);   // N=256
  dim3 gSI(CC/16,   BC/128);   // N=64
  dim3 gP (PP/16,   BC/128);   // N=96

  init_kimax_kernel<<<1, 64, 0, stream>>>(kimax);
  conv_hlast_kernel<<<(BB*CC*DD+255)/256, 256, 0, stream>>>(x_enc, Wconv, bconv, hl);

  // cross-channel attention
  wmma_gemm_kernel<DD, 0><<<gDD, tpb, 0, stream>>>(hl, Wq, bq, qb, DD);
  wmma_gemm_kernel<DD, 0><<<gDD, tpb, 0, stream>>>(hl, Wk, bk, kb, DD);
  wmma_gemm_kernel<DD, 0><<<gDD, tpb, 0, stream>>>(hl, Wv, bv, vb, DD);
  xattn_kernel<<<BB*HH, 64, 0, stream>>>(qb, kb, vb, xo);
  wmma_gemm_kernel<DD, 0><<<gDD, tpb, 0, stream>>>(xo, Wo, bo, catt, DD);

  // routing MLP + channel-selection scores
  concat_kernel<<<(BC*2*DD+255)/256, 256, 0, stream>>>(hl, catt, inp);
  wmma_gemm_kernel<2*DD, 1><<<gDD, tpb, 0, stream>>>(inp, Wm1, bm1, m1, DD);
  ki_kernel<<<(BC+255)/256, 256, 0, stream>>>(m1, Wm2, bm2, kimax);
  wmma_gemm_kernel<2*DD, 2><<<gSI, tpb, 0, stream>>>(inp, Wc, bc, si, CC);
  topk_kernel<<<BC, 64, 0, stream>>>(si, idxb);

  // gathered-token attention: project once, gather rows
  wmma_gemm_kernel<DD, 0><<<gDD, tpb, 0, stream>>>(hl, Wlq, blq, lqb, DD);
  wmma_gemm_kernel<DD, 0><<<gDD, tpb, 0, stream>>>(hl, Wlk, blk_b, hkb, DD);
  wmma_gemm_kernel<DD, 0><<<gDD, tpb, 0, stream>>>(hl, Wlv, blv, hvb, DD);
  gather_attn_kernel<<<BC, 64, 0, stream>>>(lqb, hkb, hvb, idxb, kimax, a2);
  wmma_gemm_kernel<DD, 0><<<gDD, tpb, 0, stream>>>(a2, Wlo, blo, at0, DD);

  // residual + LN + FFN + LN + projection + mean
  ln_res_kernel<<<BC, DD, 0, stream>>>(hl, at0, g1, be1, t0);
  wmma_gemm_kernel<DD, 1><<<gFF, tpb, 0, stream>>>(t0, Wf1, bf1, f1, DFFD);
  wmma_gemm_kernel<DFFD, 0><<<gDD, tpb, 0, stream>>>(f1, Wf2, bf2, f2, DD);
  ln_res_kernel<<<BC, DD, 0, stream>>>(t0, f2, g2, be2, zb);
  wmma_gemm_kernel<DD, 0><<<gP, tpb, 0, stream>>>(zb, Wp, bp, yb, PP);
  mean_kernel<<<(BB*PP+255)/256, 256, 0, stream>>>(yb, out);
}
